// MoEBlock_644245095084
// MI455X (gfx1250) — compile-verified, matrix operand order checked by reading, not run on hardware
//
#include <hip/hip_runtime.h>
#include <hip/hip_bf16.h>
#include <math.h>

#define NUM_EXPERTS 8
#define DIM 1024
#define HIDDEN 4096
#define T_TOKENS 8192   // 4 * 2048
#define MTILE 32        // tokens per workgroup (2 x 16-row WMMA tiles)
#define HCHUNK 512      // hidden chunk held in LDS between GEMM1 and GEMM2
#define XLD (DIM + 8)      // 1032 halfs -> row stride 2064B (bank-friendly)
#define HLD (HCHUNK + 8)   // 520 halfs  -> row stride 1040B

typedef _Float16 half_t;
typedef __attribute__((ext_vector_type(16))) _Float16 v16h;
typedef __attribute__((ext_vector_type(8)))  _Float16 v8h;
typedef __attribute__((ext_vector_type(8)))  float    v8f;

// Load 16 contiguous halfs (32B) as one WMMA operand slice; 16B-aligned sources.
__device__ __forceinline__ v16h ld_frag16(const half_t* p) {
    const v8h* q = (const v8h*)p;
    v8h lo = q[0];
    v8h hi = q[1];
    return __builtin_shufflevector(lo, hi, 0, 1, 2, 3, 4, 5, 6, 7,
                                           8, 9, 10, 11, 12, 13, 14, 15);
}

__device__ __forceinline__ v8f wmma_f16(v16h a, v16h b, v8f c) {
    return __builtin_amdgcn_wmma_f32_16x16x32_f16(false, a, false, b,
                                                  (short)0, c, false, false);
}

// silu(h) = h * sigmoid(h), using the HW approximate reciprocal (v_rcp_f32).
__device__ __forceinline__ float fast_silu(float h) {
    return h * __builtin_amdgcn_rcpf(1.0f + __expf(-h));
}

// ---------------------------------------------------------------------------
// Pack fp32 weights [R, K] (row-major; R = E*N, expert boundary 16-aligned)
// into f16 WMMA B-fragment order: [n_tile][k_tile][lane][16 halfs].
// Fragment (ng, kt): lane l holds row ng*16 + (l&15), cols kt*32 + (l>>4)*16 ..+16.
// A wave's B-frag load then becomes one contiguous, coalesced 1KB read.
// ---------------------------------------------------------------------------
__global__ void pack_wmma_b_kernel(const float* __restrict__ in,
                                   half_t* __restrict__ out, long R, long K) {
    const long total = (R * K) >> 4;            // number of 16-half groups
    const long ktiles = K >> 5;
    long g = (long)blockIdx.x * blockDim.x + threadIdx.x;
    const long stride = (long)gridDim.x * blockDim.x;
    for (; g < total; g += stride) {
        const long f = g >> 5;                  // fragment index
        const int lane = (int)(g & 31);
        const long ng = f / ktiles;
        const long kt = f - ng * ktiles;
        const long row = ng * 16 + (lane & 15);
        const long col = (kt << 5) + ((lane >> 4) << 4);
        const float* src = in + row * K + col;
        half_t* dst = out + (g << 4);
        for (int j = 0; j < 16; ++j) dst[j] = (half_t)src[j];
    }
}

// ---------------------------------------------------------------------------
// Router: logits = x . router_w^T ; softmax ; top-2 ; renormalize ;
// scatter into dense combine[T, E] (mirrors reference exactly).
// ---------------------------------------------------------------------------
__global__ __launch_bounds__(256)
void router_kernel(const float* __restrict__ x, const float* __restrict__ rw,
                   float* __restrict__ combine) {
    __shared__ float logits[NUM_EXPERTS];
    const int t = blockIdx.x;
    const int wave = threadIdx.x >> 5;
    const int lane = threadIdx.x & 31;
    const float* xt = x + (size_t)t * DIM;
    const float* we = rw + (size_t)wave * DIM;
    float s = 0.0f;
    for (int k = lane; k < DIM; k += 32) s += xt[k] * we[k];
    for (int off = 16; off > 0; off >>= 1) s += __shfl_xor(s, off, 32);
    if (lane == 0) logits[wave] = s;
    __syncthreads();
    if (threadIdx.x == 0) {
        float l[NUM_EXPERTS], m = -INFINITY;
        for (int e = 0; e < NUM_EXPERTS; ++e) { l[e] = logits[e]; m = fmaxf(m, l[e]); }
        float p[NUM_EXPERTS];
        for (int e = 0; e < NUM_EXPERTS; ++e) p[e] = __expf(l[e] - m);
        int i0 = 0;
        for (int e = 1; e < NUM_EXPERTS; ++e) if (p[e] > p[i0]) i0 = e;
        int i1 = (i0 == 0) ? 1 : 0;
        for (int e = 0; e < NUM_EXPERTS; ++e)
            if (e != i0 && p[e] > p[i1]) i1 = e;
        const float denom = p[i0] + p[i1];
        float o[NUM_EXPERTS];
        for (int e = 0; e < NUM_EXPERTS; ++e) o[e] = 0.0f;
        o[i0] = p[i0] / denom;
        o[i1] = p[i1] / denom;
        for (int e = 0; e < NUM_EXPERTS; ++e)
            combine[(size_t)t * NUM_EXPERTS + e] = o[e];
    }
}

// ---------------------------------------------------------------------------
// Fused dense MoE MLP, 32-token tile per workgroup (256 threads = 8 wave32s).
// Loop experts -> hidden chunks: GEMM1 -> silu * combine -> LDS f16 -> GEMM2
// accumulates into 16 persistent f32 WMMA accumulators per wave.
// Both GEMMs are k-outer: 2 LDS A-frag loads per k-step amortized over
// 4 (GEMM1) / 8 (GEMM2) packed-B streams; each B frag feeds two WMMAs.
// #pragma unroll 1 on the k-loops bounds register lookahead (no spills).
// ---------------------------------------------------------------------------
__global__ __launch_bounds__(256, 1)
void moe_fused_kernel(const float* __restrict__ x,
                      const float* __restrict__ combine,
                      const half_t* __restrict__ w1p,
                      const half_t* __restrict__ w2p,
                      float* __restrict__ out) {
    __shared__ __align__(32) half_t xtile[MTILE * XLD];   // 66.0 KB
    __shared__ __align__(32) half_t htile[MTILE * HLD];   // 33.3 KB
    __shared__ float cw_lds[MTILE * NUM_EXPERTS];         // 1 KB

    const int t0 = blockIdx.x * MTILE;
    const int tid = threadIdx.x;
    const int wave = tid >> 5;        // 0..7
    const int lane = tid & 31;
    const int r16 = lane & 15;
    const int m_hi = lane >> 4;       // fragment half selector

    // Stage x tile: fp32 global -> f16 LDS
    for (int i = tid; i < MTILE * DIM; i += 256) {
        const int r = i >> 10;          // / DIM
        const int c = i & (DIM - 1);
        xtile[r * XLD + c] = (half_t)x[(size_t)(t0 + r) * DIM + c];
    }
    cw_lds[tid] = combine[(size_t)t0 * NUM_EXPERTS + tid];   // 256 == MTILE*E
    __syncthreads();

    // Persistent accumulators: this wave owns out cols [wave*128, +128)
    v8f oacc[8][2];
    #pragma unroll
    for (int nt = 0; nt < 8; ++nt) { oacc[nt][0] = (v8f){}; oacc[nt][1] = (v8f){}; }

    const half_t* a0base = xtile + r16 * XLD + (m_hi << 4);
    const half_t* a1base = xtile + (16 + r16) * XLD + (m_hi << 4);
    const half_t* h0base = htile + r16 * HLD + (m_hi << 4);
    const half_t* h1base = htile + (16 + r16) * HLD + (m_hi << 4);
    // per-lane component of packed-B addresses (scalar part added per access)
    const half_t* w1lane = w1p + (size_t)lane * 16;
    const half_t* w2lane = w2p + (size_t)lane * 16;

    for (int e = 0; e < NUM_EXPERTS; ++e) {
        for (int c = 0; c < HIDDEN / HCHUNK; ++c) {
            // ---- GEMM1: this wave computes hidden cols [wave*64, +64) of chunk
            v8f acc[4][2];
            #pragma unroll
            for (int nt = 0; nt < 4; ++nt) { acc[nt][0] = (v8f){}; acc[nt][1] = (v8f){}; }

            // fragment-stream scalar offsets (halfs): frag index * 512
            const size_t ng0 =
                (size_t)e * (HIDDEN / 16) + ((c * HCHUNK + wave * 64) >> 4);
            #pragma unroll 1
            for (int k0 = 0; k0 < DIM; k0 += 32) {
                const v16h A0 = ld_frag16(a0base + k0);
                const v16h A1 = ld_frag16(a1base + k0);
                const size_t kt = (size_t)k0 >> 5;
                #pragma unroll
                for (int nt = 0; nt < 4; ++nt) {
                    const size_t soff = ((ng0 + nt) * (DIM / 32) + kt) * 512;
                    const v16h b = ld_frag16(w1lane + soff);
                    acc[nt][0] = wmma_f16(A0, b, acc[nt][0]);
                    acc[nt][1] = wmma_f16(A1, b, acc[nt][1]);
                }
            }
            // silu * combine weight, scatter into htile (f16)
            #pragma unroll
            for (int nt = 0; nt < 4; ++nt) {
                const int hc = wave * 64 + nt * 16 + r16;
                #pragma unroll
                for (int v = 0; v < 8; ++v) {
                    const float c0 = cw_lds[(v + 8 * m_hi) * NUM_EXPERTS + e];
                    const float c1 = cw_lds[(16 + v + 8 * m_hi) * NUM_EXPERTS + e];
                    const float h0 = fast_silu(acc[nt][0][v]) * c0;
                    htile[(v + 8 * m_hi) * HLD + hc] = (half_t)h0;
                    const float h1 = fast_silu(acc[nt][1][v]) * c1;
                    htile[(16 + v + 8 * m_hi) * HLD + hc] = (half_t)h1;
                }
            }
            __syncthreads();

            // ---- GEMM2 partial: out cols [wave*128, +128), K = this chunk.
            const size_t ng2base = (size_t)e * (DIM / 16) + (wave * 8);
            #pragma unroll 1
            for (int k0 = 0; k0 < HCHUNK; k0 += 32) {
                const v16h A0 = ld_frag16(h0base + k0);
                const v16h A1 = ld_frag16(h1base + k0);
                const size_t kt = (size_t)(c * HCHUNK + k0) >> 5;
                #pragma unroll
                for (int nt = 0; nt < 8; ++nt) {
                    const size_t soff =
                        ((ng2base + nt) * (HIDDEN / 32) + kt) * 512;
                    const v16h b = ld_frag16(w2lane + soff);
                    oacc[nt][0] = wmma_f16(A0, b, oacc[nt][0]);
                    oacc[nt][1] = wmma_f16(A1, b, oacc[nt][1]);
                }
            }
            __syncthreads();   // protect htile before next chunk overwrites it
        }
    }

    // Write fp32 output (coalesced across lanes)
    for (int nt = 0; nt < 8; ++nt) {
        const int ncol = wave * 128 + nt * 16 + r16;
        for (int mt = 0; mt < 2; ++mt)
            for (int v = 0; v < 8; ++v) {
                const int m = mt * 16 + v + 8 * m_hi;
                out[(size_t)(t0 + m) * DIM + ncol] = oacc[nt][mt][v];
            }
    }
}

// ---------------------------------------------------------------------------
extern "C" void kernel_launch(void* const* d_in, const int* in_sizes, int n_in,
                              void* d_out, int out_size, void* d_ws, size_t ws_size,
                              hipStream_t stream) {
    (void)in_sizes; (void)n_in; (void)out_size; (void)ws_size;
    const float* x        = (const float*)d_in[0];  // [4,2048,1024]
    const float* router_w = (const float*)d_in[1];  // [8,1024]
    const float* w1       = (const float*)d_in[2];  // [8,4096,1024]
    const float* w2       = (const float*)d_in[3];  // [8,1024,4096]
    float* out = (float*)d_out;

    // Workspace layout:
    //   combine : T*E f32            =   256 KB
    //   w1p     : E*H*D f16 (packed) =  67.1 MB
    //   w2p     : E*D*H f16 (packed) =  67.1 MB
    char* ws = (char*)d_ws;
    float* combine = (float*)ws;
    half_t* w1p = (half_t*)(ws + (256 << 10));
    half_t* w2p = w1p + (size_t)NUM_EXPERTS * HIDDEN * DIM;

    pack_wmma_b_kernel<<<4096, 256, 0, stream>>>(
        w1, w1p, (long)NUM_EXPERTS * HIDDEN, (long)DIM);
    pack_wmma_b_kernel<<<4096, 256, 0, stream>>>(
        w2, w2p, (long)NUM_EXPERTS * DIM, (long)HIDDEN);
    router_kernel<<<T_TOKENS, 256, 0, stream>>>(x, router_w, combine);
    moe_fused_kernel<<<T_TOKENS / MTILE, 256, 0, stream>>>(
        x, combine, w1p, w2p, out);
}
